// GapAwareAttention_25812753449151
// MI455X (gfx1250) — compile-verified, matrix operand order checked by reading, not run on hardware
//
#include <hip/hip_runtime.h>
#include <hip/hip_bf16.h>
#include <math.h>

// GapAwareAttention for MI455X (gfx1250), wave32 + WMMA bf16 + async-to-LDS.
//
// Pipeline:
//   0) memset d_out (attended accumulator)
//   p) prep: x -> bf16 [N][128]; Wq/Wk/Wv/Wo -> bf16 transposed [n][k]
//   1) qkv_gemm_k : q,k,v = x@W + b     (v_wmma_f32_16x16x32_bf16,
//                                        global_load_async_to_lds_b128 staging)
//   2) edge_scores_k : per-(edge,head) q[dst].k[src]/4 + block max partials
//   3) reduce_max_k  : global per-head max (softmax over ALL edges)
//   4) exp_k         : scores = exp(s - max), block sum partials
//   5) reduce_sum_k  : per-head 1/denom
//   6) scatter_k     : attended[dst] += w * v[src]  (global_atomic_add_f32)
//   7) out_gemm_k    : out = x + attended@Wo + bo   (WMMA, in-place tile)
//
// Workspace ≈ 234 MB: q,k,v (3*N*128 f32), scores (E*8 f32), partials,
//                     x_bf16 (N*128), Wt_bf16 (4*128*128).

#define N_NODES 100000
#define E_EDGES 1600000
#define HCH 128
#define NHEADS 8
#define HD 16
#define LDSTRIDE 136   // bf16 elements per LDS row (pad 128 -> 136: lanes hit distinct banks)

#define USE_ASYNC_TLDS 1

typedef __bf16 v16bf __attribute__((ext_vector_type(16)));
typedef float  v8f   __attribute__((ext_vector_type(8)));
typedef __attribute__((address_space(3))) unsigned short lds_ushort_t;

union Frag16 {
    v16bf    v;
    unsigned u[8];
};

__device__ __forceinline__ unsigned short f32_to_bf16(float x) {
    unsigned a = __float_as_uint(x);
    a = (a + 0x7FFFu + ((a >> 16) & 1u)) >> 16;   // round-to-nearest-even
    return (unsigned short)a;
}

__device__ __forceinline__ unsigned pack2_bf16(float lo, float hi) {
    return (unsigned)f32_to_bf16(lo) | ((unsigned)f32_to_bf16(hi) << 16);
}

// A-fragments for 16x16x32 bf16 from the LDS tile, as 2x b128 per k-chunk.
// Lane l: row m=l&15, kb=(l>>4)*8; u[0..3] = K kc*32+kb..+7, u[4..7] = +16.
__device__ __forceinline__ void load_a_frags(const unsigned short* xt, int lane,
                                             Frag16 af[4]) {
    const int m  = lane & 15;
    const int kb = (lane >> 4) * 8;
    #pragma unroll
    for (int kc = 0; kc < 4; ++kc) {
        const uint4* p = (const uint4*)(xt + m * LDSTRIDE + kc * 32 + kb);
        uint4 lo = p[0];          // 8 bf16, K = kc*32+kb .. +7
        uint4 hi = p[2];          // 8 bf16, K = +16
        af[kc].u[0] = lo.x; af[kc].u[1] = lo.y; af[kc].u[2] = lo.z; af[kc].u[3] = lo.w;
        af[kc].u[4] = hi.x; af[kc].u[5] = hi.y; af[kc].u[6] = hi.z; af[kc].u[7] = hi.w;
    }
}

// B-fragment from K-major bf16 weights Wt[n][k]: 2x global b128 per k-chunk.
__device__ __forceinline__ void load_b_frag(const unsigned short* __restrict__ Wt,
                                            int col, int kc, int kb, Frag16& bf) {
    const uint4* p = (const uint4*)(Wt + (size_t)col * HCH + kc * 32 + kb);
    uint4 lo = p[0];
    uint4 hi = p[2];
    bf.u[0] = lo.x; bf.u[1] = lo.y; bf.u[2] = lo.z; bf.u[3] = lo.w;
    bf.u[4] = hi.x; bf.u[5] = hi.y; bf.u[6] = hi.z; bf.u[7] = hi.w;
}

// ---------------- prep: x -> bf16 ----------------
__global__ __launch_bounds__(256) void cvt_x_k(const float* __restrict__ x,
                                               unsigned short* __restrict__ xb) {
    size_t base = ((size_t)blockIdx.x * 256 + threadIdx.x) * 8;
    const float4* p = (const float4*)(x + base);
    float4 a = p[0], b = p[1];
    uint4 o;
    o.x = pack2_bf16(a.x, a.y);
    o.y = pack2_bf16(a.z, a.w);
    o.z = pack2_bf16(b.x, b.y);
    o.w = pack2_bf16(b.z, b.w);
    *(uint4*)(xb + base) = o;
}

// ---------------- prep: W[k][n] f32 -> Wt[mtx][n][k] bf16 ----------------
__global__ __launch_bounds__(256) void cvt_w_k(
    const float* __restrict__ W0, const float* __restrict__ W1,
    const float* __restrict__ W2, const float* __restrict__ W3,
    unsigned short* __restrict__ Wt) {
    int t   = blockIdx.x * 256 + threadIdx.x;       // 0..65535
    int mtx = t >> 14;
    int idx = t & 16383;
    int n = idx >> 7, k = idx & 127;
    const float* W = (mtx == 0) ? W0 : (mtx == 1) ? W1 : (mtx == 2) ? W2 : W3;
    Wt[(size_t)mtx * 16384 + (size_t)n * HCH + k] = f32_to_bf16(W[(size_t)k * HCH + n]);
}

// Stage a 16x128 bf16 tile (4 KB) into the padded LDS tile.
__device__ __forceinline__ void stage_tile_bf16(const unsigned short* __restrict__ xb,
                                                int tileBase, unsigned short* xt,
                                                int tid) {
    const int row = tid >> 4;
    const int col = (tid & 15) * 8;                 // 8 bf16 = 16 bytes per thread
    const unsigned short* g = xb + (size_t)(tileBase + row) * HCH + col;
#if USE_ASYNC_TLDS
    unsigned long long ga = (unsigned long long)(const void*)g;
    unsigned loff = (unsigned)(size_t)(lds_ushort_t*)(xt + row * LDSTRIDE + col);
    // ASYNCcnt-tracked async copy: global -> LDS, 16 bytes/lane.
    asm volatile("global_load_async_to_lds_b128 %0, %1, off"
                 :: "v"(loff), "v"(ga) : "memory");
    asm volatile("s_wait_asynccnt 0" ::: "memory");
#else
    *(uint4*)(xt + row * LDSTRIDE + col) = *(const uint4*)g;
#endif
}

// ---------------- 1) fused QKV GEMM ----------------
__global__ __launch_bounds__(256) void qkv_gemm_k(
    const unsigned short* __restrict__ xb, const unsigned short* __restrict__ Wt,
    const float* __restrict__ bq, const float* __restrict__ bk,
    const float* __restrict__ bv,
    float* __restrict__ q, float* __restrict__ k, float* __restrict__ v) {
    __shared__ __align__(16) unsigned short xt[16 * LDSTRIDE];

    const int tid = threadIdx.x;
    const int tileBase = blockIdx.x * 16;

    stage_tile_bf16(xb, tileBase, xt, tid);
    __syncthreads();

    const int lane   = tid & 31;
    const int wv32   = tid >> 5;             // wave id 0..7 -> column tile
    const int kb     = (lane >> 4) * 8;
    const int col    = wv32 * 16 + (lane & 15);
    const int rowOff = (lane >> 4) * 8;

    Frag16 af[4];
    load_a_frags(xt, lane, af);

    const float* bs[3] = {bq, bk, bv};
    float*       os[3] = {q, k, v};

    #pragma unroll
    for (int mtx = 0; mtx < 3; ++mtx) {
        const unsigned short* W = Wt + (size_t)mtx * 16384;
        float bias = bs[mtx][col];
        v8f c;
        #pragma unroll
        for (int r = 0; r < 8; ++r) c[r] = bias;

        #pragma unroll
        for (int kc = 0; kc < 4; ++kc) {
            Frag16 bf;
            load_b_frag(W, col, kc, kb, bf);
            c = __builtin_amdgcn_wmma_f32_16x16x32_bf16(
                    false, af[kc].v, false, bf.v, (short)0, c, false, false);
        }

        float* o = os[mtx];
        #pragma unroll
        for (int r = 0; r < 8; ++r)
            o[(size_t)(tileBase + r + rowOff) * HCH + col] = c[r];
    }
}

// ---------------- 2) edge scores + block max ----------------
__global__ __launch_bounds__(256) void edge_scores_k(
    const float* __restrict__ q, const float* __restrict__ k,
    const long long* __restrict__ eidx, float* __restrict__ sc,
    float* __restrict__ pmax) {
    __shared__ float red[256];
    const int t = threadIdx.x;
    const int h = t & 7;
    const size_t e = (size_t)blockIdx.x * 32 + (t >> 3);

    const int src = (int)eidx[e];
    const int dst = (int)eidx[(size_t)E_EDGES + e];

    const float4* qp = (const float4*)(q + (size_t)dst * HCH + h * HD);
    const float4* kp = (const float4*)(k + (size_t)src * HCH + h * HD);
    float s = 0.f;
    #pragma unroll
    for (int i = 0; i < 4; ++i) {
        float4 a = qp[i], b = kp[i];
        s += a.x * b.x + a.y * b.y + a.z * b.z + a.w * b.w;
    }
    s *= 0.25f;                       // 1/sqrt(HD)
    sc[e * NHEADS + h] = s;

    red[t] = s;
    __syncthreads();
    if (t < 8) {
        float m = -3.4e38f;
        #pragma unroll 8
        for (int i = 0; i < 32; ++i) m = fmaxf(m, red[i * 8 + t]);
        pmax[(size_t)blockIdx.x * 8 + t] = m;
    }
}

// ---------------- 3/5) tiny per-head reductions ----------------
__global__ __launch_bounds__(256) void reduce_max_k(
    const float* __restrict__ part, int nblk, float* __restrict__ gmax) {
    __shared__ float red[256];
    const int t = threadIdx.x, h = t & 7, j0 = t >> 3;
    float m = -3.4e38f;
    for (int j = j0; j < nblk; j += 32) m = fmaxf(m, part[(size_t)j * 8 + h]);
    red[t] = m;
    __syncthreads();
    if (t < 8) {
        float mm = -3.4e38f;
        for (int i = 0; i < 32; ++i) mm = fmaxf(mm, red[i * 8 + t]);
        gmax[t] = mm;
    }
}

__global__ __launch_bounds__(256) void reduce_sum_k(
    const float* __restrict__ part, int nblk, float* __restrict__ ginv) {
    __shared__ float red[256];
    const int t = threadIdx.x, h = t & 7, j0 = t >> 3;
    float s = 0.f;
    for (int j = j0; j < nblk; j += 32) s += part[(size_t)j * 8 + h];
    red[t] = s;
    __syncthreads();
    if (t < 8) {
        float ss = 0.f;
        for (int i = 0; i < 32; ++i) ss += red[i * 8 + t];
        ginv[t] = 1.0f / ss;
    }
}

// ---------------- 4) exp(s - max) + block sums ----------------
__global__ __launch_bounds__(256) void exp_k(
    float* __restrict__ sc, const float* __restrict__ gmax,
    float* __restrict__ psum) {
    __shared__ float red[256];
    const int t = threadIdx.x;
    const int h = t & 7;                     // 256 % 8 == 0
    const size_t i = (size_t)blockIdx.x * 256 + t;
    float val = __expf(sc[i] - gmax[h]);
    sc[i] = val;
    red[t] = val;
    __syncthreads();
    if (t < 8) {
        float s = 0.f;
        #pragma unroll 8
        for (int j = 0; j < 32; ++j) s += red[j * 8 + t];
        psum[(size_t)blockIdx.x * 8 + t] = s;
    }
}

// ---------------- 6) weighted scatter-add ----------------
__global__ __launch_bounds__(256) void scatter_k(
    const float* __restrict__ sc, const float* __restrict__ ginv,
    const float* __restrict__ v, const long long* __restrict__ eidx,
    float* __restrict__ att) {
    const size_t t = (size_t)blockIdx.x * 256 + threadIdx.x;
    const int h = (int)(t & 7);
    const size_t e = t >> 3;
    const float w = sc[t] * ginv[h];
    const int src = (int)eidx[e];
    const int dst = (int)eidx[(size_t)E_EDGES + e];

    const float4* vp = (const float4*)(v + (size_t)src * HCH + h * HD);
    float* ap = att + (size_t)dst * HCH + h * HD;
    #pragma unroll
    for (int i = 0; i < 4; ++i) {
        float4 vv = vp[i];
        atomicAdd(ap + i * 4 + 0, w * vv.x);
        atomicAdd(ap + i * 4 + 1, w * vv.y);
        atomicAdd(ap + i * 4 + 2, w * vv.z);
        atomicAdd(ap + i * 4 + 3, w * vv.w);
    }
}

// ---------------- 7) output projection + residual ----------------
__global__ __launch_bounds__(256) void out_gemm_k(
    const float* __restrict__ x, const unsigned short* __restrict__ Wto,
    const float* __restrict__ bo,
    float* __restrict__ out /* holds attended (f32) on entry */) {
    __shared__ __align__(16) unsigned short at[16 * LDSTRIDE];

    const int tid = threadIdx.x;
    const int tileBase = blockIdx.x * 16;

    // Stage + convert attended tile (f32 -> bf16), 8 elements/thread.
    {
        const int row = tid >> 4;
        const int col = (tid & 15) * 8;
        const float4* p = (const float4*)(out + (size_t)(tileBase + row) * HCH + col);
        float4 a = p[0], b = p[1];
        uint4 o;
        o.x = pack2_bf16(a.x, a.y);
        o.y = pack2_bf16(a.z, a.w);
        o.z = pack2_bf16(b.x, b.y);
        o.w = pack2_bf16(b.z, b.w);
        *(uint4*)(at + row * LDSTRIDE + col) = o;
    }
    __syncthreads();   // everyone has read this tile before any wave overwrites it

    const int lane   = tid & 31;
    const int wv32   = tid >> 5;
    const int kb     = (lane >> 4) * 8;
    const int col    = wv32 * 16 + (lane & 15);
    const int rowOff = (lane >> 4) * 8;

    Frag16 af[4];
    load_a_frags(at, lane, af);

    float bias = bo[col];
    v8f c;
    #pragma unroll
    for (int r = 0; r < 8; ++r) c[r] = bias;

    #pragma unroll
    for (int kc = 0; kc < 4; ++kc) {
        Frag16 bf;
        load_b_frag(Wto, col, kc, kb, bf);
        c = __builtin_amdgcn_wmma_f32_16x16x32_bf16(
                false, af[kc].v, false, bf.v, (short)0, c, false, false);
    }

    #pragma unroll
    for (int r = 0; r < 8; ++r) {
        size_t idx = (size_t)(tileBase + r + rowOff) * HCH + col;
        out[idx] = c[r] + x[idx];
    }
}

// ---------------- launcher ----------------
extern "C" void kernel_launch(void* const* d_in, const int* in_sizes, int n_in,
                              void* d_out, int out_size, void* d_ws, size_t ws_size,
                              hipStream_t stream) {
    const float*     x    = (const float*)d_in[0];
    const long long* eidx = (const long long*)d_in[1];
    const float* Wq = (const float*)d_in[2]; const float* bq = (const float*)d_in[3];
    const float* Wk = (const float*)d_in[4]; const float* bk = (const float*)d_in[5];
    const float* Wv = (const float*)d_in[6]; const float* bv = (const float*)d_in[7];
    const float* Wo = (const float*)d_in[8]; const float* bo = (const float*)d_in[9];
    float* out = (float*)d_out;

    const int nEBlk = E_EDGES / 32;              // 50000 (edge-score blocks)
    const int nXBlk = (E_EDGES * NHEADS) / 256;  // 50000 (exp / scatter blocks)

    // Workspace layout. f32 region first, then 16B-aligned bf16 region.
    float* ws   = (float*)d_ws;
    float* q    = ws;
    float* k    = q  + (size_t)N_NODES * HCH;
    float* v    = k  + (size_t)N_NODES * HCH;
    float* sc   = v  + (size_t)N_NODES * HCH;          // E*8
    float* pmax = sc + (size_t)E_EDGES * NHEADS;       // nEBlk*8
    float* psum = pmax + (size_t)nEBlk * 8;            // nXBlk*8
    float* gmax = psum + (size_t)nXBlk * 8;            // 8
    float* ginv = gmax + 8;                            // 8
    unsigned short* xb = (unsigned short*)(ginv + 8);  // N*128 bf16 (offset 16B-aligned)
    unsigned short* Wt = xb + (size_t)N_NODES * HCH;   // 4*128*128 bf16 [mtx][n][k]

    // attended accumulator lives in d_out; zero it.
    hipMemsetAsync(d_out, 0, (size_t)N_NODES * HCH * sizeof(float), stream);

    cvt_x_k<<<(N_NODES * HCH) / (256 * 8), 256, 0, stream>>>(x, xb);
    cvt_w_k<<<(4 * HCH * HCH) / 256, 256, 0, stream>>>(Wq, Wk, Wv, Wo, Wt);

    qkv_gemm_k<<<N_NODES / 16, 256, 0, stream>>>(xb, Wt, bq, bk, bv, q, k, v);
    edge_scores_k<<<nEBlk, 256, 0, stream>>>(q, k, eidx, sc, pmax);
    reduce_max_k<<<1, 256, 0, stream>>>(pmax, nEBlk, gmax);
    exp_k<<<nXBlk, 256, 0, stream>>>(sc, gmax, psum);
    reduce_sum_k<<<1, 256, 0, stream>>>(psum, nXBlk, ginv);
    scatter_k<<<nXBlk, 256, 0, stream>>>(sc, ginv, v, eidx, out);
    out_gemm_k<<<N_NODES / 16, 256, 0, stream>>>(x, Wt + 3 * 16384, bo, out);
}
